// GraphCPI_GIN_36850819400363
// MI455X (gfx1250) — compile-verified, matrix operand order checked by reading, not run on hardware
//
#include <hip/hip_runtime.h>
#include <hip/hip_bf16.h>

// ---------------- model dimensions ----------------
#define NNODES 32768
#define NEDGES 131072
#define NGRAPH 1024
#define FEAT   78
#define DIM    32
#define DOUT   128
#define EMBD   100
#define SEQ    1000
#define NF     32
#define KW     8
#define CLEN   93
#define CFLAT  (NF*CLEN)   // 2976

// conv tiling
#define TGROUP 32          // t values per block (4 waves x 8)
#define TPERW  8           // t values per wave
#define SPAN   40          // staged emb elements per row
#define RS     44          // LDS row stride (elements) -> 88B, conflict-free

typedef __bf16        v16bf __attribute__((ext_vector_type(16)));
typedef float         v8f   __attribute__((ext_vector_type(8)));
typedef unsigned int  u32x4 __attribute__((ext_vector_type(4)));

union BFrag { v16bf v; unsigned short u[16]; u32x4 q[2]; };

__device__ inline unsigned short f2bf(float x) {
  unsigned int u = __builtin_bit_cast(unsigned int, x);
  return (unsigned short)((u + 0x7FFFu + ((u >> 16) & 1u)) >> 16);
}

// Flat LDS addresses carry the LDS offset in addr[31:0] (ISA 10.2 aperture rules),
// so truncating a generic pointer to shared memory yields the wave-relative offset.
__device__ inline unsigned ldsOff(const void* p) {
  return (unsigned)(unsigned long long)p;
}

// ASYNCcnt-tracked DMA of 8 bytes global -> LDS, per lane.
__device__ inline void asyncLoadB64(unsigned lds, const void* gaddr) {
  asm volatile("global_load_async_to_lds_b64 %0, %1, off"
               :: "v"(lds), "v"((unsigned long long)gaddr)
               : "memory");
}
__device__ inline void waitAsync0() {
  asm volatile("s_wait_asynccnt 0x0" ::: "memory");
}

// ---------------- utility kernels ----------------
__global__ void k_zero(float* p, int n) {
  int i = blockIdx.x * blockDim.x + threadIdx.x;
  if (i < n) p[i] = 0.0f;
}

__global__ void k_cvt_bf16(const float* __restrict__ s, unsigned short* __restrict__ d, int n) {
  int i = blockIdx.x * blockDim.x + threadIdx.x;
  if (i < n) d[i] = f2bf(s[i]);
}

// ---------------- GIN: edge scatter-add ----------------
__global__ void k_scatter(const float* __restrict__ h, const int* __restrict__ ei,
                          float* __restrict__ msg, int fi) {
  int i = blockIdx.x * blockDim.x + threadIdx.x;
  if (i >= NEDGES * fi) return;
  int e = i / fi;
  int d = i - e * fi;
  int s = ei[e];            // src row
  int t = ei[NEDGES + e];   // dst row
  unsafeAtomicAdd(&msg[t * fi + d], h[s * fi + d]);
}

// ---------------- GIN: MLP + stats ----------------
template<int FI>
__global__ void k_gin_mlp(const float* __restrict__ hin, const float* __restrict__ msg,
                          const float* __restrict__ W1, const float* __restrict__ b1,
                          const float* __restrict__ W2, const float* __restrict__ b2,
                          float* __restrict__ z, float* __restrict__ stats) {
  __shared__ float sW1[FI * DIM];
  __shared__ float sW2[DIM * DIM];
  __shared__ float sb1[DIM], sb2[DIM];
  __shared__ float ssum[DIM], ssq[DIM];
  int tid = threadIdx.x;
  for (int i = tid; i < FI * DIM; i += blockDim.x) sW1[i] = W1[i];
  for (int i = tid; i < DIM * DIM; i += blockDim.x) sW2[i] = W2[i];
  if (tid < DIM) { sb1[tid] = b1[tid]; sb2[tid] = b2[tid]; ssum[tid] = 0.f; ssq[tid] = 0.f; }
  __syncthreads();

  int node = blockIdx.x * blockDim.x + tid;   // grid sized exactly
  float zin[FI];
#pragma unroll
  for (int d = 0; d < FI; ++d) zin[d] = hin[node * FI + d] + msg[node * FI + d];
  float a[DIM];
#pragma unroll
  for (int o = 0; o < DIM; ++o) {
    float s = sb1[o];
#pragma unroll
    for (int d = 0; d < FI; ++d) s = fmaf(zin[d], sW1[d * DIM + o], s);
    a[o] = fmaxf(s, 0.0f);
  }
#pragma unroll
  for (int o = 0; o < DIM; ++o) {
    float s = sb2[o];
#pragma unroll
    for (int d = 0; d < DIM; ++d) s = fmaf(a[d], sW2[d * DIM + o], s);
    s = fmaxf(s, 0.0f);
    z[node * DIM + o] = s;
    atomicAdd(&ssum[o], s);        // ds_add_f32
    atomicAdd(&ssq[o], s * s);
  }
  __syncthreads();
  if (tid < DIM) {
    unsafeAtomicAdd(&stats[tid], ssum[tid]);
    unsafeAtomicAdd(&stats[DIM + tid], ssq[tid]);
  }
}

// ---------------- GIN: batchnorm apply ----------------
__global__ void k_bn(const float* __restrict__ z, const float* __restrict__ stats,
                     const float* __restrict__ gamma, const float* __restrict__ beta,
                     float* __restrict__ h) {
  int i = blockIdx.x * blockDim.x + threadIdx.x;   // exact grid
  int d = i & (DIM - 1);
  float mu = stats[d] * (1.0f / NNODES);
  float var = stats[DIM + d] * (1.0f / NNODES) - mu * mu;
  float sc = rsqrtf(var + 1e-5f) * gamma[d];
  h[i] = (z[i] - mu) * sc + beta[d];
}

// ---------------- graph pooling ----------------
__global__ void k_pool(const float* __restrict__ h, const int* __restrict__ batch,
                       float* __restrict__ pooled) {
  int i = blockIdx.x * blockDim.x + threadIdx.x;   // exact grid NNODES*DIM
  int n = i >> 5;
  int d = i & 31;
  unsafeAtomicAdd(&pooled[batch[n] * DIM + d], h[i]);
}

// ---------------- drug head ----------------
__global__ void k_drug(const float* __restrict__ pooled, const float* __restrict__ Wxd,
                       const float* __restrict__ bxd, unsigned short* __restrict__ hidbf) {
  int i = blockIdx.x * blockDim.x + threadIdx.x;   // exact grid NGRAPH*DOUT
  int b = i >> 7;
  int o = i & (DOUT - 1);
  float s = bxd[o];
#pragma unroll
  for (int d = 0; d < DIM; ++d) s = fmaf(pooled[b * DIM + d], Wxd[d * DOUT + o], s);
  hidbf[b * 256 + o] = f2bf(fmaxf(s, 0.0f));
}

// ---------------- protein conv via WMMA bf16 + async LDS staging ----------------
// out[b,f,t] = sum_{s,k} emb[target[b,s]][t+k] * w[f,s,k]
// Block: 4 waves, 16 b-rows, 32 t values (8 per wave). Double-buffered async emb tile.
__global__ void __launch_bounds__(128)
k_conv_wmma(const unsigned short* __restrict__ ebf,
            const unsigned short* __restrict__ wbf,
            const int* __restrict__ target,
            const float* __restrict__ conv_b,
            unsigned short* __restrict__ cbf) {
  __shared__ unsigned short tile[2][64 * RS];

  int tid = threadIdx.x;
  int lane = tid & 31;
  int wave = tid >> 5;
  int btile = blockIdx.x * 16;
  int t0 = blockIdx.y * TGROUP;           // 0, 32, 64
  int t0w = t0 + wave * TPERW;
  int count = CLEN - t0w; if (count > TPERW) count = TPERW;  // always >= 5
  int half = lane >> 4;
  int mn = lane & 15;

  v8f acc0[TPERW], acc1[TPERW];
#pragma unroll
  for (int j = 0; j < TPERW; ++j)
#pragma unroll
    for (int i = 0; i < 8; ++i) { acc0[j][i] = 0.f; acc1[j][i] = 0.f; }

  // Stage 64 rows (16 b x 4 s) of SPAN=40 emb elements for chunk sb=4*ci.
  // 640 b64 transfers per tile, 5 per thread; src 8B-aligned (tok*200 + 2*t0 + 8j).
  auto stage = [&](int ci, int buf) {
    int sb = ci * 4;
#pragma unroll
    for (int i = 0; i < 5; ++i) {
      int c = tid + 128 * i;          // 0..639
      int r = c / 10;                 // row 0..63  (r = s4*16 + bi)
      int j = c - r * 10;             // b64 chunk 0..9
      int s4 = r >> 4, bi = r & 15;
      int tok = target[(btile + bi) * SEQ + sb + s4];
      const unsigned short* src = ebf + tok * EMBD + t0 + j * 4;
      asyncLoadB64(ldsOff(&tile[buf][r * RS + j * 4]), src);
    }
  };

  stage(0, 0);
  for (int ci = 0; ci < SEQ / 4; ++ci) {
    waitAsync0();
    __syncthreads();
    if (ci + 1 < SEQ / 4) stage(ci + 1, (ci + 1) & 1);

    const unsigned short* tl = &tile[ci & 1][0];
    // 16-element sliding windows for this wave's 8 t values, aligned ds_load_b64 x4
    union { unsigned long long q[4]; unsigned short u[16]; } w0, w1;
    {
      const unsigned long long* p0 =
          (const unsigned long long*)(tl + (half * 16 + mn) * RS + wave * TPERW);
      const unsigned long long* p1 =
          (const unsigned long long*)(tl + ((2 + half) * 16 + mn) * RS + wave * TPERW);
#pragma unroll
      for (int k = 0; k < 4; ++k) { w0.q[k] = p0[k]; w1.q[k] = p1[k]; }
    }

    // B fragments (L2-resident weights), loaded once per chunk, reused by 16 WMMAs
    int sb = ci * 4;
    BFrag bb0, bb1;
    bb0.q[0] = *(const u32x4*)(wbf + mn * (SEQ * KW) + (sb + half) * KW);
    bb0.q[1] = *(const u32x4*)(wbf + mn * (SEQ * KW) + (sb + 2 + half) * KW);
    bb1.q[0] = *(const u32x4*)(wbf + (16 + mn) * (SEQ * KW) + (sb + half) * KW);
    bb1.q[1] = *(const u32x4*)(wbf + (16 + mn) * (SEQ * KW) + (sb + 2 + half) * KW);

#pragma unroll
    for (int j = 0; j < TPERW; ++j) {
      if (j >= count) break;          // wave-uniform
      BFrag a;
#pragma unroll
      for (int k = 0; k < 8; ++k) { a.u[k] = w0.u[j + k]; a.u[8 + k] = w1.u[j + k]; }
      acc0[j] = __builtin_amdgcn_wmma_f32_16x16x32_bf16(false, a.v, false, bb0.v,
                                                        (short)0, acc0[j], false, false);
      acc1[j] = __builtin_amdgcn_wmma_f32_16x16x32_bf16(false, a.v, false, bb1.v,
                                                        (short)0, acc1[j], false, false);
    }
  }

  // epilogue: D element r -> row = r + 8*half, col = mn; layout c[b][f*93+t] (+bias) bf16
#pragma unroll
  for (int j = 0; j < TPERW; ++j) {
    if (j >= count) break;
    int t = t0w + j;
#pragma unroll
    for (int r = 0; r < 8; ++r) {
      int brow = btile + r + 8 * half;
      cbf[brow * CFLAT + mn * CLEN + t]        = f2bf(acc0[j][r] + conv_b[mn]);
      cbf[brow * CFLAT + (16 + mn) * CLEN + t] = f2bf(acc1[j][r] + conv_b[16 + mn]);
    }
  }
}

// ---------------- generic bf16 WMMA GEMM ----------------
template<int RELU, int OUTBF>
__global__ void k_gemm_wmma(const unsigned short* __restrict__ A,
                            const unsigned short* __restrict__ B,
                            const float* __restrict__ bias, void* __restrict__ outp,
                            int M, int N, int K, int ldo, int outOff, int totalTiles) {
  int lane = threadIdx.x & 31;
  int wave = threadIdx.x >> 5;
  int tile = blockIdx.x * (blockDim.x >> 5) + wave;
  if (tile >= totalTiles) return;             // wave-uniform
  int tilesN = N >> 4;
  int tM = tile / tilesN, tN = tile - tM * tilesN;
  int half = lane >> 4;
  int mn = lane & 15;
  int m = tM * 16 + mn;
  int n = tN * 16 + mn;

  v8f acc;
#pragma unroll
  for (int i = 0; i < 8; ++i) acc[i] = 0.f;

  for (int kk = 0; kk < K; kk += 32) {
    BFrag a, bmat;
    const u32x4* pa = (const u32x4*)(A + (size_t)m * K + kk + 8 * half); // 16B aligned (K%8==0)
    a.q[0] = pa[0];
    a.q[1] = pa[2];                           // +16 bf16 elements
#pragma unroll
    for (int i = 0; i < 16; ++i) {
      int kidx = kk + i + (i & 8) + 8 * half;
      bmat.u[i] = B[(size_t)kidx * N + n];
    }
    acc = __builtin_amdgcn_wmma_f32_16x16x32_bf16(false, a.v, false, bmat.v,
                                                  (short)0, acc, false, false);
  }
#pragma unroll
  for (int r = 0; r < 8; ++r) {
    int mr = tM * 16 + r + 8 * half;
    float v = acc[r] + bias[n];
    if (RELU) v = fmaxf(v, 0.0f);
    if (OUTBF) ((unsigned short*)outp)[(size_t)mr * ldo + outOff + n] = f2bf(v);
    else       ((float*)outp)[(size_t)mr * ldo + outOff + n] = v;
  }
}

// ---------------- fc3 ----------------
__global__ void k_fc3(const float* __restrict__ h2, const float* __restrict__ W,
                      const float* __restrict__ b, float* __restrict__ out) {
  int i = blockIdx.x * blockDim.x + threadIdx.x;
  if (i >= NGRAPH) return;
  float s = b[0];
  for (int d = 0; d < 512; ++d) s = fmaf(h2[i * 512 + d], W[d], s);
  out[i] = s;
}

// ---------------- host launch ----------------
extern "C" void kernel_launch(void* const* d_in, const int* in_sizes, int n_in,
                              void* d_out, int out_size, void* d_ws, size_t ws_size,
                              hipStream_t stream) {
  const float* x      = (const float*)d_in[0];
  const int*   ei     = (const int*)d_in[1];
  const int*   batch  = (const int*)d_in[2];
  const int*   target = (const int*)d_in[3];
  const float* Wxd  = (const float*)d_in[34];
  const float* bxd  = (const float*)d_in[35];
  const float* emb  = (const float*)d_in[36];
  const float* cw   = (const float*)d_in[37];
  const float* cb   = (const float*)d_in[38];
  const float* Wxt  = (const float*)d_in[39];
  const float* bxt  = (const float*)d_in[40];
  const float* Wf1  = (const float*)d_in[41];
  const float* bf1  = (const float*)d_in[42];
  const float* Wf2  = (const float*)d_in[43];
  const float* bf2  = (const float*)d_in[44];
  const float* Wf3  = (const float*)d_in[45];
  const float* bf3  = (const float*)d_in[46];

  char* ws = (char*)d_ws;
  size_t off = 0;
  auto alloc = [&](size_t bytes) { size_t r = off; off += (bytes + 255) & ~(size_t)255; return r; };
  float*          msg    = (float*)(ws + alloc((size_t)NNODES * FEAT * 4));
  float*          h      = (float*)(ws + alloc((size_t)NNODES * DIM * 4));
  float*          z      = (float*)(ws + alloc((size_t)NNODES * DIM * 4));
  float*          stats  = (float*)(ws + alloc(64 * 4));
  float*          pooled = (float*)(ws + alloc((size_t)NGRAPH * DIM * 4));
  unsigned short* hidbf  = (unsigned short*)(ws + alloc((size_t)NGRAPH * 256 * 2));
  unsigned short* ebf    = (unsigned short*)(ws + alloc((size_t)8000 * EMBD * 2 + 256));
  unsigned short* wcbf   = (unsigned short*)(ws + alloc((size_t)NF * SEQ * KW * 2));
  unsigned short* cbf    = (unsigned short*)(ws + alloc((size_t)NGRAPH * CFLAT * 2));
  unsigned short* wxtbf  = (unsigned short*)(ws + alloc((size_t)CFLAT * DOUT * 2));
  unsigned short* wf1bf  = (unsigned short*)(ws + alloc((size_t)256 * 1024 * 2));
  unsigned short* wf2bf  = (unsigned short*)(ws + alloc((size_t)1024 * 512 * 2));
  unsigned short* h1bf   = (unsigned short*)(ws + alloc((size_t)NGRAPH * 1024 * 2));
  float*          h2     = (float*)(ws + alloc((size_t)NGRAPH * 512 * 4));

  auto gridFor = [](long n) { return (unsigned)((n + 255) / 256); };

  // weight/table conversions to bf16
  k_cvt_bf16<<<gridFor(8000L * EMBD), 256, 0, stream>>>(emb, ebf, 8000 * EMBD);
  k_cvt_bf16<<<gridFor((long)NF * SEQ * KW), 256, 0, stream>>>(cw, wcbf, NF * SEQ * KW);
  k_cvt_bf16<<<gridFor((long)CFLAT * DOUT), 256, 0, stream>>>(Wxt, wxtbf, CFLAT * DOUT);
  k_cvt_bf16<<<gridFor(256L * 1024), 256, 0, stream>>>(Wf1, wf1bf, 256 * 1024);
  k_cvt_bf16<<<gridFor(1024L * 512), 256, 0, stream>>>(Wf2, wf2bf, 1024 * 512);

  // ---- 5 GIN layers ----
  const float* hin = x;
  for (int L = 0; L < 5; ++L) {
    int fi = (L == 0) ? FEAT : DIM;
    const float* W1 = (const float*)d_in[4 + 6 * L];
    const float* b1 = (const float*)d_in[5 + 6 * L];
    const float* W2 = (const float*)d_in[6 + 6 * L];
    const float* b2 = (const float*)d_in[7 + 6 * L];
    const float* gm = (const float*)d_in[8 + 6 * L];
    const float* bt = (const float*)d_in[9 + 6 * L];

    k_zero<<<gridFor((long)NNODES * fi), 256, 0, stream>>>(msg, NNODES * fi);
    k_scatter<<<gridFor((long)NEDGES * fi), 256, 0, stream>>>(hin, ei, msg, fi);
    k_zero<<<1, 64, 0, stream>>>(stats, 64);
    if (L == 0)
      k_gin_mlp<FEAT><<<NNODES / 128, 128, 0, stream>>>(hin, msg, W1, b1, W2, b2, z, stats);
    else
      k_gin_mlp<DIM><<<NNODES / 128, 128, 0, stream>>>(hin, msg, W1, b1, W2, b2, z, stats);
    k_bn<<<(NNODES * DIM) / 256, 256, 0, stream>>>(z, stats, gm, bt, h);
    hin = h;
  }

  // ---- pooling + drug head ----
  k_zero<<<gridFor((long)NGRAPH * DIM), 256, 0, stream>>>(pooled, NGRAPH * DIM);
  k_pool<<<(NNODES * DIM) / 256, 256, 0, stream>>>(h, batch, pooled);
  k_drug<<<(NGRAPH * DOUT) / 256, 256, 0, stream>>>(pooled, Wxd, bxd, hidbf);

  // ---- protein conv (WMMA + async LDS staging): 64 b-tiles x 3 t-groups ----
  k_conv_wmma<<<dim3(NGRAPH / 16, 3), 128, 0, stream>>>(ebf, wcbf, target, cb, cbf);

  // ---- prot = c @ Wxt + bxt -> hid[:, 128:256] ----
  {
    int tiles = (NGRAPH / 16) * (DOUT / 16);   // 512
    k_gemm_wmma<0, 1><<<tiles / 8, 256, 0, stream>>>(cbf, wxtbf, bxt, hidbf,
                                                     NGRAPH, DOUT, CFLAT, 256, 128, tiles);
  }
  // ---- fc1 ----
  {
    int tiles = (NGRAPH / 16) * (1024 / 16);   // 4096
    k_gemm_wmma<1, 1><<<tiles / 8, 256, 0, stream>>>(hidbf, wf1bf, bf1, h1bf,
                                                     NGRAPH, 1024, 256, 1024, 0, tiles);
  }
  // ---- fc2 ----
  {
    int tiles = (NGRAPH / 16) * (512 / 16);    // 2048
    k_gemm_wmma<1, 0><<<tiles / 8, 256, 0, stream>>>(h1bf, wf2bf, bf2, h2,
                                                     NGRAPH, 512, 1024, 512, 0, tiles);
  }
  // ---- fc3 -> d_out ----
  k_fc3<<<4, 256, 0, stream>>>(h2, Wf3, bf3, (float*)d_out);
}
